// GlobalAttention_33105607917733
// MI455X (gfx1250) — matrix operand order repaired and checked by reading, not compiled, not run on hardware
//
#include <hip/hip_runtime.h>
#include <hip/hip_bf16.h>
#include <math.h>

typedef _Float16 f16;
typedef __attribute__((ext_vector_type(16))) _Float16 v16h;
typedef __attribute__((ext_vector_type(8)))  _Float16 v8h;
typedef __attribute__((ext_vector_type(8)))  float    v8f;

// Problem constants (fixed by the reference)
constexpr int Bc = 4;          // batch
constexpr int Tn = 8192;       // sequence length (multiple of W -> no padding)
constexpr int Cc = 512;        // d_model
constexpr int Wd = 64;         // dilation window
constexpr int Sd = 128;        // sub-sequence length = T / W
constexpr int Nn = Bc * Wd;    // 256 sub-sequences
constexpr int Hh = 8;          // heads
constexpr int Dh = 64;         // head dim
constexpr int Mr = Nn * Sd;    // 32768 total rows

// ---------------------------------------------------------------------------
// WMMA helpers (gfx1250, wave32).  V_WMMA_F32_16X16X32_F16: D = A(16x32) x
// B(32x16) + C(16x16 f32).
// ---------------------------------------------------------------------------
__device__ inline v8f wmma_f16(v16h a, v16h b, v8f c) {
  return __builtin_amdgcn_wmma_f32_16x16x32_f16(
      false, a, false, b, (short)0, c, false, false);
}

__device__ inline v16h cat16(v8h lo, v8h hi) {
  return __builtin_shufflevector(lo, hi,
      0, 1, 2, 3, 4, 5, 6, 7, 8, 9, 10, 11, 12, 13, 14, 15);
}

// A fragment (16x32) from row-major f16 (global or LDS), leading dim lda.
// Lane l: m = l&15; halves 0..15 = K (l>>4)*16 + 0..15.
__device__ inline v16h load_a_f16(const f16* A, int lda) {
  const int l = threadIdx.x & 31;
  const f16* p = A + (size_t)(l & 15) * lda + ((l >> 4) << 4);
  return cat16(*(const v8h*)(p), *(const v8h*)(p + 8));
}

// B fragment for a TRANSPOSED operand: logical B[k][n] = Kp[n*ldk + k]
// (q @ k^T): per-lane contiguous 32B -> 2x global_load_b128.
__device__ inline v16h load_bT_f16(const f16* Kp, int ldk) {
  const int l = threadIdx.x & 31;
  const f16* p = Kp + (size_t)(l & 15) * ldk + ((l >> 4) << 4);
  return cat16(*(const v8h*)(p), *(const v8h*)(p + 8));
}

// B fragment from memory pre-swizzled into fragment order:
// base[lane*16 + j] = B[kB(lane)+j][n(lane)]  -> 2x b128 per lane.
__device__ inline v16h load_b_swz(const f16* base) {
  const int l = threadIdx.x & 31;
  const f16* p = base + ((size_t)l << 4);
  return cat16(*(const v8h*)(p), *(const v8h*)(p + 8));
}

// ---------------------------------------------------------------------------
// Kernel 0a: swizzle one 512x512 fp32 weight matrix into fragment-ordered f16:
// dst[((kc*8 + hg)*4 + t)*512 + lane*16 + j] =
//   W[(kc*32 + (lane>>4)*16 + j) * 512 + hg*64 + t*16 + (lane&15)]
// grid = (128, 4): x = kc*8+hg tile, y = which matrix.  Via LDS so both global
// reads (float4) and writes (b128) are wide and coalesced.
// ---------------------------------------------------------------------------
__global__ void __launch_bounds__(256)
ga_prep_swz(const float* __restrict__ Wq, const float* __restrict__ Wk,
            const float* __restrict__ Wv, const float* __restrict__ Wo,
            f16* wq, f16* wk, f16* wv, f16* wo) {
  const float* src;
  f16* dst;
  switch (blockIdx.y) {
    case 0:  src = Wq; dst = wq; break;
    case 1:  src = Wk; dst = wk; break;
    case 2:  src = Wv; dst = wv; break;
    default: src = Wo; dst = wo; break;
  }
  const int tile = blockIdx.x;      // kc*8 + hg
  const int kc = tile >> 3;
  const int hg = tile & 7;

  __shared__ f16 T[32 * 64];        // row-major [K=32][N=64] tile

  // coalesced fp32 tile load, convert to f16
  {
    int e = threadIdx.x * 8;        // 2048 / 256 = 8 elements per thread
    int kk = e >> 6, jj = e & 63;
    const float* sp = src + (size_t)(kc * 32 + kk) * Cc + hg * 64 + jj;
    float4 f0 = *(const float4*)(sp + 0);
    float4 f1 = *(const float4*)(sp + 4);
    T[e + 0] = (f16)f0.x; T[e + 1] = (f16)f0.y; T[e + 2] = (f16)f0.z; T[e + 3] = (f16)f0.w;
    T[e + 4] = (f16)f1.x; T[e + 5] = (f16)f1.y; T[e + 6] = (f16)f1.z; T[e + 7] = (f16)f1.w;
  }
  __syncthreads();

  // fragment-ordered write: 8 consecutive dst elements per thread (b128 store)
  {
    int o = threadIdx.x * 8;
    int j0 = o & 15;                // 0 or 8
    int l = (o >> 4) & 31;
    int t = (o >> 9) & 3;
    int kB = (l >> 4) << 4;
    int nlow = l & 15;
    v8h tv;
#pragma unroll
    for (int j = 0; j < 8; ++j)
      tv[j] = T[(kB + j0 + j) * 64 + t * 16 + nlow];
    *(v8h*)(dst + (size_t)tile * 2048 + o) = tv;
  }
}

// Kernel 0b: RoPE tables.
__global__ void ga_rope(float* cosT, float* sinT) {
  int i = blockIdx.x * 256 + threadIdx.x;
  if (i < Sd * Dh) {
    int s = i >> 6;
    int d = i & 63;
    float invf = __powf(10000.0f, -(float)(d & 31) / 32.0f);
    float ang = (float)s * invf;
    cosT[i] = __cosf(ang);
    sinT[i] = __sinf(ang);
  }
}

// ---------------------------------------------------------------------------
// Kernel 1: fused windowed-gather + Q/K/V projection + RoPE (single x pass).
// grid = (Nn, Hh), block 256 (8 waves).  No LDS, no barriers: B fragments are
// 2x global_load_b128 from the pre-swizzled weights (L2-resident, shared by
// all 2048 workgroups).  Outputs [N, H, S, Dh] f16.
// ---------------------------------------------------------------------------
__global__ void __launch_bounds__(256)
ga_qkv(const float* __restrict__ x,
       const f16* __restrict__ wq, const f16* __restrict__ wk,
       const f16* __restrict__ wv,
       f16* __restrict__ q16, f16* __restrict__ k16, f16* __restrict__ v16,
       const float* __restrict__ cosT, const float* __restrict__ sinT) {
  const int n = blockIdx.x;
  const int h = blockIdx.y;
  const int wave = threadIdx.x >> 5;
  const int l = threadIdx.x & 31;

  // Row gather: global row r = n*S + s maps to x[b, s*W + wd, :]
  const int b = n >> 6;
  const int wd = n & 63;
  const int s_a = wave * 16 + (l & 15);
  const float* rowp = x + ((size_t)b * Tn + (size_t)s_a * Wd + wd) * Cc;

  v8f aq[4], ak[4], av[4];
#pragma unroll
  for (int t = 0; t < 4; ++t) { aq[t] = (v8f){}; ak[t] = (v8f){}; av[t] = (v8f){}; }

  const int kA = (l >> 4) * 16;
  for (int kc = 0; kc < 16; ++kc) {
    // A fragment straight from x (fp32 -> f16 on load, float4 reads)
    const float* p = rowp + kc * 32 + kA;
    float4 f0 = *(const float4*)(p + 0);
    float4 f1 = *(const float4*)(p + 4);
    float4 f2 = *(const float4*)(p + 8);
    float4 f3 = *(const float4*)(p + 12);
    v16h a;
    a[0] = (f16)f0.x;  a[1] = (f16)f0.y;  a[2]  = (f16)f0.z;  a[3]  = (f16)f0.w;
    a[4] = (f16)f1.x;  a[5] = (f16)f1.y;  a[6]  = (f16)f1.z;  a[7]  = (f16)f1.w;
    a[8] = (f16)f2.x;  a[9] = (f16)f2.y;  a[10] = (f16)f2.z;  a[11] = (f16)f2.w;
    a[12] = (f16)f3.x; a[13] = (f16)f3.y; a[14] = (f16)f3.z;  a[15] = (f16)f3.w;

    const size_t fb = ((size_t)(kc * 8 + h) * 4) << 9;  // *4 frags * 512
#pragma unroll
    for (int t = 0; t < 4; ++t) {
      aq[t] = wmma_f16(a, load_b_swz(wq + fb + ((size_t)t << 9)), aq[t]);
      ak[t] = wmma_f16(a, load_b_swz(wk + fb + ((size_t)t << 9)), ak[t]);
      av[t] = wmma_f16(a, load_b_swz(wv + fb + ((size_t)t << 9)), av[t]);
    }
  }

  // Epilogue: D layout row = i + 8*(l>>4), col = (l&15) + 16*t
  const int half = l >> 4;
  const int nn = l & 15;
  const size_t headbase = ((size_t)(n * Hh + h)) * Sd * Dh;

  // RoPE for q and k: pair column d (t in {0,1}) with d+32 (t+2)
#pragma unroll
  for (int t = 0; t < 2; ++t) {
#pragma unroll
    for (int i = 0; i < 8; ++i) {
      int s_i = wave * 16 + i + 8 * half;
      int d = nn + 16 * t;
      float cs = cosT[s_i * Dh + d];
      float sn = sinT[s_i * Dh + d];
      size_t base = headbase + (size_t)s_i * Dh;
      float q0 = aq[t][i], q2 = aq[t + 2][i];
      q16[base + d]      = (f16)(q0 * cs - q2 * sn);
      q16[base + d + 32] = (f16)(q2 * cs + q0 * sn);
      float k0v = ak[t][i], k2v = ak[t + 2][i];
      k16[base + d]      = (f16)(k0v * cs - k2v * sn);
      k16[base + d + 32] = (f16)(k2v * cs + k0v * sn);
    }
  }
  // v: direct store
#pragma unroll
  for (int t = 0; t < 4; ++t) {
#pragma unroll
    for (int i = 0; i < 8; ++i) {
      int s_i = wave * 16 + i + 8 * half;
      v16[headbase + (size_t)s_i * Dh + nn + 16 * t] = (f16)av[t][i];
    }
  }
}

// ---------------------------------------------------------------------------
// Kernel 2: attention for one (n, h) per workgroup.  8 waves; wave w owns
// query rows w*16..w*16+15.  Output ao16 is [M, C] f16 for the Wo GEMM.
// ---------------------------------------------------------------------------
__global__ void __launch_bounds__(256)
ga_attn(const f16* __restrict__ q16, const f16* __restrict__ k16,
        const f16* __restrict__ v16, const unsigned char* __restrict__ pmask,
        f16* __restrict__ ao16) {
  const int n = blockIdx.x;
  const int h = blockIdx.y;
  const size_t hb = ((size_t)(n * Hh + h)) * Sd * Dh;
  const f16* qh = q16 + hb;
  const f16* kh = k16 + hb;
  const f16* vh = v16 + hb;

  __shared__ f16 Vs[Sd * Dh];   // 16 KB: V, fragment-swizzled (16 frags)
  __shared__ f16 As[Sd * Sd];   // 32 KB: attention probabilities, row-major

  // swizzled V fill: vectorized b128 source reads, b16 scatter to LDS
  for (int e = threadIdx.x * 8; e < Sd * Dh; e += 256 * 8) {
    v8h vv = *(const v8h*)(vh + e);
    int row = e >> 6;
    int col0 = e & 63;            // multiple of 8
    int rem = row & 31;
    int frag = ((row >> 5) << 2) | (col0 >> 4);
    int j = rem & 15;
    int lh = (rem >> 4) << 4;
#pragma unroll
    for (int u = 0; u < 8; ++u)
      Vs[(((frag << 5) | lh | ((col0 + u) & 15)) << 4) | j] = vv[u];
  }

  const int wave = threadIdx.x >> 5;
  const int l = threadIdx.x & 31;
  const int qr0 = wave * 16;
  const int half = l >> 4;
  const int nn = l & 15;

  // Q fragments for this wave's 16 rows (K = head dim, two 32-chunks)
  v16h aq0 = load_a_f16(qh + (size_t)qr0 * Dh, Dh);
  v16h aq1 = load_a_f16(qh + (size_t)qr0 * Dh + 32, Dh);

  // scores: 16 x 128 per wave, 8 column tiles
  v8f sc[8];
#pragma unroll
  for (int ct = 0; ct < 8; ++ct) {
    v16h b0 = load_bT_f16(kh + (size_t)(ct * 16) * Dh, Dh);
    v16h b1 = load_bT_f16(kh + (size_t)(ct * 16) * Dh + 32, Dh);
    v8f c = (v8f){};
    c = wmma_f16(aq0, b0, c);
    c = wmma_f16(aq1, b1, c);
    sc[ct] = c;
  }

  // scale + padding mask (pk[n, key] = padding_mask[b, key*W + wd])
  const float scale = 0.125f;  // 1/sqrt(64)
  const int b = n >> 6;
  const int wd = n & 63;
#pragma unroll
  for (int ct = 0; ct < 8; ++ct) {
    int key = ct * 16 + nn;
    bool msk = pmask[(size_t)b * Tn + (size_t)key * Wd + wd] != 0;
#pragma unroll
    for (int i = 0; i < 8; ++i)
      sc[ct][i] = msk ? -1.0e30f : sc[ct][i] * scale;
  }

  // softmax: each 16-lane half shares rows -> xor 1/2/4/8 reductions
  float rinv[8];
#pragma unroll
  for (int i = 0; i < 8; ++i) {
    float mx = sc[0][i];
#pragma unroll
    for (int ct = 1; ct < 8; ++ct) mx = fmaxf(mx, sc[ct][i]);
    mx = fmaxf(mx, __shfl_xor(mx, 1));
    mx = fmaxf(mx, __shfl_xor(mx, 2));
    mx = fmaxf(mx, __shfl_xor(mx, 4));
    mx = fmaxf(mx, __shfl_xor(mx, 8));
    float sum = 0.0f;
#pragma unroll
    for (int ct = 0; ct < 8; ++ct) {
      float e = __expf(sc[ct][i] - mx);
      sc[ct][i] = e;
      sum += e;
    }
    sum += __shfl_xor(sum, 1);
    sum += __shfl_xor(sum, 2);
    sum += __shfl_xor(sum, 4);
    sum += __shfl_xor(sum, 8);
    rinv[i] = 1.0f / sum;
  }

  __syncthreads();  // Vs ready

  // write probabilities to LDS (row-major; A-fragment reads are b128)
#pragma unroll
  for (int ct = 0; ct < 8; ++ct) {
#pragma unroll
    for (int i = 0; i < 8; ++i) {
      int row = qr0 + i + 8 * half;
      As[(size_t)row * Sd + ct * 16 + nn] = (f16)sc[ct][i];
    }
  }
  __syncthreads();

  // out = attn(16x128) @ V(128x64)
  v8f o[4];
#pragma unroll
  for (int t = 0; t < 4; ++t) o[t] = (v8f){};
#pragma unroll
  for (int kt = 0; kt < 4; ++kt) {
    v16h a = load_a_f16(As + (size_t)qr0 * Sd + kt * 32, Sd);
#pragma unroll
    for (int dt = 0; dt < 4; ++dt)
      o[dt] = wmma_f16(a, load_b_swz(Vs + ((((kt << 2) | dt) * 32) << 4)), o[dt]);
  }

  // normalize by row sum and store to [M, C] at column h*64
#pragma unroll
  for (int i = 0; i < 8; ++i) {
    int row = n * Sd + qr0 + i + 8 * half;
    float rv = rinv[i];
#pragma unroll
    for (int dt = 0; dt < 4; ++dt)
      ao16[(size_t)row * Cc + h * Dh + dt * 16 + nn] = (f16)(o[dt][i] * rv);
  }
}

// ---------------------------------------------------------------------------
// Kernel 3: output projection (attn_out @ Wo) + inverse window scatter.
// grid = (Nn, 8) block 256.  Barrier-free; B from pre-swizzled wo.
// ---------------------------------------------------------------------------
__global__ void __launch_bounds__(256)
ga_oproj(const f16* __restrict__ ao16, const f16* __restrict__ wo,
         float* __restrict__ out) {
  const int n = blockIdx.x;
  const int hg = blockIdx.y;
  const int wave = threadIdx.x >> 5;
  const int l = threadIdx.x & 31;

  const size_t r0 = (size_t)n * Sd + wave * 16;
  v8f acc[4];
#pragma unroll
  for (int t = 0; t < 4; ++t) acc[t] = (v8f){};

  for (int kc = 0; kc < 16; ++kc) {
    v16h a = load_a_f16(ao16 + r0 * Cc + kc * 32, Cc);
    const size_t fb = ((size_t)(kc * 8 + hg) * 4) << 9;
#pragma unroll
    for (int t = 0; t < 4; ++t)
      acc[t] = wmma_f16(a, load_b_swz(wo + fb + ((size_t)t << 9)), acc[t]);
  }

  // scatter back: row r = n*S + s  ->  out[b, s*W + wd, :]
  const int b = n >> 6;
  const int wd = n & 63;
  const int half = l >> 4;
  const int nn = l & 15;
  const int j0 = hg * 64;
#pragma unroll
  for (int i = 0; i < 8; ++i) {
    int s = wave * 16 + i + 8 * half;
    size_t obase = ((size_t)b * Tn + (size_t)s * Wd + wd) * Cc + j0;
#pragma unroll
    for (int t = 0; t < 4; ++t)
      out[obase + t * 16 + nn] = acc[t][i];
  }
}

// ---------------------------------------------------------------------------
extern "C" void kernel_launch(void* const* d_in, const int* in_sizes, int n_in,
                              void* d_out, int out_size, void* d_ws, size_t ws_size,
                              hipStream_t stream) {
  const float* x = (const float*)d_in[0];
  const unsigned char* pmask = (const unsigned char*)d_in[1];
  const float* Wq = (const float*)d_in[2];
  const float* Wk = (const float*)d_in[3];
  const float* Wv = (const float*)d_in[4];
  const float* Wo = (const float*)d_in[5];
  float* out = (float*)d_out;

  // workspace carve-up
  char* ws = (char*)d_ws;
  const size_t MC2 = (size_t)Mr * Cc * sizeof(f16);   // 32 MB each
  f16* q16 = (f16*)ws;            ws += MC2;
  f16* k16 = (f16*)ws;            ws += MC2;
  f16* v16 = (f16*)ws;            ws += MC2;
  f16* ao16 = (f16*)ws;           ws += MC2;
  f16* wq16 = (f16*)ws;           ws += (size_t)Cc * Cc * sizeof(f16);
  f16* wk16 = (f16*)ws;           ws += (size_t)Cc * Cc * sizeof(f16);
  f16* wv16 = (f16*)ws;           ws += (size_t)Cc * Cc * sizeof(f16);
  f16* wo16 = (f16*)ws;           ws += (size_t)Cc * Cc * sizeof(f16);
  float* cosT = (float*)ws;       ws += (size_t)Sd * Dh * sizeof(float);
  float* sinT = (float*)ws;       ws += (size_t)Sd * Dh * sizeof(float);

  ga_prep_swz<<<dim3(128, 4), 256, 0, stream>>>(
      Wq, Wk, Wv, Wo, wq16, wk16, wv16, wo16);
  ga_rope<<<32, 256, 0, stream>>>(cosT, sinT);

  ga_qkv<<<dim3(Nn, Hh), 256, 0, stream>>>(
      x, wq16, wk16, wv16, q16, k16, v16, cosT, sinT);

  ga_attn<<<dim3(Nn, Hh), 256, 0, stream>>>(q16, k16, v16, pmask, ao16);

  ga_oproj<<<dim3(Nn, 8), 256, 0, stream>>>(ao16, wo16, out);
}
